// EquivariantSwarmDecoder_23450521436537
// MI455X (gfx1250) — matrix operand order repaired and checked by reading, not compiled, not run on hardware
//
#include <hip/hip_runtime.h>
#include <hip/hip_bf16.h>
#include <cmath>

// ---------------------------------------------------------------------------
// EquivariantSwarmDecoder for MI455X (gfx1250, wave32, WMMA).
// B=128, K=64, M=2048, C=32, QD=128, AD=128, H=256.
// ---------------------------------------------------------------------------

typedef __attribute__((ext_vector_type(16))) _Float16 v16h;
typedef __attribute__((ext_vector_type(8)))  _Float16 v8h;
typedef __attribute__((ext_vector_type(8)))  float    v8f;

union F16x16 { v16h v; v8h h[2]; };

// exact gelu (used only in tiny prep kernels)
__device__ __forceinline__ float gelu_exact(float x) {
    return 0.5f * x * (1.0f + erff(x * 0.70710678118654752f));
}

// branch-free fast gelu for the hot fused kernel: x * sigmoid(1.59577(x+0.044715x^3))
// (tanh-form gelu; deviation from exact gelu ~1e-3, below the f16 GEMM noise floor)
__device__ __forceinline__ float gelu_fast(float x) {
    float u = x * (1.5957691216057308f + 0.07135481283f * x * x);
    float e = __expf(-u);
    return x * __builtin_amdgcn_rcpf(1.0f + e);
}

#define NB      128
#define NK      64
#define NM      2048
#define NH      256
#define NROWS   (NB * NM)          // 262144
#define ROWS_PER_BLOCK 128

// ---------------------------------------------------------------------------
// Prep: f16 weight conversion, fused head matrix (48x256), v = qW^T kW, c0.
// ---------------------------------------------------------------------------
__global__ __launch_bounds__(256) void k_prep(
    const float* __restrict__ pm_W2, const float* __restrict__ pm_W3,
    const float* __restrict__ tW, const float* __restrict__ tb,
    const float* __restrict__ wW, const float* __restrict__ wb,
    const float* __restrict__ cW, const float* __restrict__ cb,
    const float* __restrict__ qW, const float* __restrict__ qb,
    const float* __restrict__ kW,
    _Float16* __restrict__ W2h, _Float16* __restrict__ W3h,
    _Float16* __restrict__ headWh, float* __restrict__ headB,
    float* __restrict__ vvec, float* __restrict__ c0)
{
    int i = blockIdx.x * 256 + threadIdx.x;
    if (i < NH * NH) { W2h[i] = (_Float16)pm_W2[i]; W3h[i] = (_Float16)pm_W3[i]; }
    if (i < 48 * NH) {
        int r = i >> 8, c = i & 255;
        float w = 0.0f;
        if (r < 32)       w = tW[r * NH + c];
        else if (r == 32) w = wW[c];
        else if (r == 33) w = cW[c];
        headWh[i] = (_Float16)w;
    }
    if (i < 48) {
        float b = 0.0f;
        if (i < 32) b = tb[i]; else if (i == 32) b = wb[0]; else if (i == 33) b = cb[0];
        headB[i] = b;
    }
    if (i < 128) {              // v[j] = sum_d qW[d][j] * kW[d]
        float acc = 0.0f;
        for (int d = 0; d < 128; ++d) acc += qW[d * 128 + i] * kW[d];
        vvec[i] = acc;
    }
    if (i == 0) {               // c0 = qb . kW  (constant, cancels in softmax but kept)
        float acc = 0.0f;
        for (int d = 0; d < 128; ++d) acc += qb[d] * kW[d];
        *c0 = acc;
    }
}

// ---------------------------------------------------------------------------
// inv[b,k] = ||g[b,k,:]||
// ---------------------------------------------------------------------------
__global__ __launch_bounds__(256) void k_inv(const float* __restrict__ g,
                                             float* __restrict__ inv)
{
    int i = blockIdx.x * 256 + threadIdx.x;
    if (i < NB * NK) {
        float x = g[i * 3 + 0], y = g[i * 3 + 1], z = g[i * 3 + 2];
        inv[i] = sqrtf(x * x + y * y + z * z);
    }
}

// ---------------------------------------------------------------------------
// qdot[m] = (query[m] . v + c0) / sqrt(128)
// ---------------------------------------------------------------------------
__global__ __launch_bounds__(256) void k_qdot(const float* __restrict__ query,
                                              const float* __restrict__ vvec,
                                              const float* __restrict__ c0,
                                              float* __restrict__ qdot)
{
    __shared__ float sv[128];
    int t = threadIdx.x;
    if (t < 128) sv[t] = vvec[t];
    __syncthreads();
    int m = blockIdx.x * 256 + t;
    if (m < NM) {
        float acc = *c0;
        for (int j = 0; j < 128; ++j) acc += query[m * 128 + j] * sv[j];
        qdot[m] = acc * 0.08838834764831845f;   // 1/sqrt(128)
    }
}

// ---------------------------------------------------------------------------
// global MLP (64->256->256->256) + fold into Gg[b] = gs @ W1g^T + pm_b1.
// One block per batch row; thread t = output neuron t.
// ---------------------------------------------------------------------------
__global__ __launch_bounds__(256) void k_gmlp(
    const float* __restrict__ inv,
    const float* __restrict__ W1, const float* __restrict__ b1,
    const float* __restrict__ W2, const float* __restrict__ b2,
    const float* __restrict__ W3, const float* __restrict__ b3,
    const float* __restrict__ pmW1, const float* __restrict__ pmb1,
    float* __restrict__ Gg)
{
    __shared__ float sa[NH], sb[NH];
    int b = blockIdx.x, t = threadIdx.x;
    if (t < NK) sa[t] = inv[b * NK + t];
    __syncthreads();
    float acc = b1[t];
    for (int j = 0; j < NK; ++j) acc += sa[j] * W1[t * NK + j];
    sb[t] = gelu_exact(acc);
    __syncthreads();
    acc = b2[t];
    for (int j = 0; j < NH; ++j) acc += sb[j] * W2[t * NH + j];
    sa[t] = gelu_exact(acc);
    __syncthreads();
    acc = b3[t];
    for (int j = 0; j < NH; ++j) acc += sa[j] * W3[t * NH + j];
    sb[t] = acc;                             // global_s[b][t]
    __syncthreads();
    acc = pmb1[t];
    for (int j = 0; j < NH; ++j) acc += sb[j] * pmW1[t * 385 + j];
    Gg[b * NH + t] = acc;
}

// ---------------------------------------------------------------------------
// Qq[m] = query[m] @ W1q^T   (W1q = pm_W1[:, 256:384])
// ---------------------------------------------------------------------------
__global__ __launch_bounds__(256) void k_qq(const float* __restrict__ query,
                                            const float* __restrict__ pmW1,
                                            float* __restrict__ Qq)
{
    __shared__ float sq[128];
    int m = blockIdx.x, t = threadIdx.x;
    if (t < 128) sq[t] = query[m * 128 + t];
    __syncthreads();
    float acc = 0.0f;
    for (int j = 0; j < 128; ++j) acc += sq[j] * pmW1[t * 385 + 256 + j];
    Qq[m * NH + t] = acc;
}

// ---------------------------------------------------------------------------
// attention: A = softmax_k(qdot[m]*inv[b,k]);  y0 = A@g;  ai = A@inv.
// 256 m per block, 8 blocks per batch row.
// ---------------------------------------------------------------------------
__global__ __launch_bounds__(256) void k_attn(const float* __restrict__ g,
                                              const float* __restrict__ inv,
                                              const float* __restrict__ qdot,
                                              float* __restrict__ ai,
                                              float* __restrict__ y0)
{
    __shared__ float sinv[NK];
    __shared__ float sg[NK * 3];
    int b = blockIdx.x >> 3, seg = blockIdx.x & 7, t = threadIdx.x;
    if (t < NK)      sinv[t] = inv[b * NK + t];
    if (t < NK * 3)  sg[t]   = g[b * NK * 3 + t];
    __syncthreads();
    int m = seg * 256 + t;
    float qd = qdot[m];
    float mx = -1e30f;
    for (int k = 0; k < NK; ++k) mx = fmaxf(mx, qd * sinv[k]);
    float s = 0.f, a0 = 0.f, a1 = 0.f, a2 = 0.f, an = 0.f;
    for (int k = 0; k < NK; ++k) {
        float e = __expf(qd * sinv[k] - mx);
        s += e;
        a0 += e * sg[k * 3 + 0];
        a1 += e * sg[k * 3 + 1];
        a2 += e * sg[k * 3 + 2];
        an += e * sinv[k];
    }
    float rs = 1.0f / s;
    int row = b * NM + m;
    y0[row * 3 + 0] = a0 * rs;
    y0[row * 3 + 1] = a1 * rs;
    y0[row * 3 + 2] = a2 * rs;
    ai[row] = an * rs;
}

// ---------------------------------------------------------------------------
// Fused point-MLP + heads. 8 waves/block, 16 rows/wave, 128 rows/block.
// layer1 elementwise (rank-structured) -> LDS f16
// layer2/3: v_wmma_f32_16x16x32_f16, weights f16 from global (L2-resident),
// activations round-trip LDS per wave (no cross-wave sharing -> no barriers).
// ---------------------------------------------------------------------------
__global__ __launch_bounds__(256) void k_fused(
    const float* __restrict__ Gg, const float* __restrict__ Qq,
    const float* __restrict__ ai, const float* __restrict__ y0,
    const float* __restrict__ pm_W1,
    const float* __restrict__ pm_b2, const float* __restrict__ pm_b3,
    const _Float16* __restrict__ W2h, const _Float16* __restrict__ W3h,
    const _Float16* __restrict__ headW, const float* __restrict__ headB,
    float* __restrict__ out_y, float* __restrict__ out_type,
    float* __restrict__ out_w)
{
    __shared__ __attribute__((aligned(16))) _Float16 sh[8][16 * NH];   // 64 KB
    const int tid  = threadIdx.x;
    const int wave = tid >> 5;
    const int lane = tid & 31;
    const int R0 = blockIdx.x * ROWS_PER_BLOCK;
    const int bb = R0 >> 11;                 // 128 rows always inside one batch
    const int rbase = R0 + wave * 16;
    _Float16* hb = sh[wave];

    // ---- layer 1: pre = Gg[b] + Qq[m] + ai*w1a  (bias folded into Gg) ----
    float ggv[8], w1av[8];
#pragma unroll
    for (int j = 0; j < 8; ++j) {
        int c = lane + j * 32;
        ggv[j]  = Gg[bb * NH + c];
        w1av[j] = pm_W1[c * 385 + 384];
    }
    for (int r = 0; r < 16; ++r) {
        int row = rbase + r;
        int m = row & (NM - 1);
        float air = ai[row];
#pragma unroll
        for (int j = 0; j < 8; ++j) {
            int c = lane + j * 32;
            float pre = ggv[j] + Qq[m * NH + c] + air * w1av[j];
            hb[r * NH + c] = (_Float16)gelu_fast(pre);
        }
    }

    const int arow = lane & 15;
    const int akb  = (lane >> 4) << 4;       // A: 16 contiguous halves
    const int bcol16 = lane & 15;
    const int bkb  = (lane >> 4) << 4;       // B: 16 contiguous halves
    const int prow = (lane >> 4) << 3;       // D: rows v + prow

    // ---- hidden layers 2 and 3 (256x256 each) ----
    for (int L = 0; L < 2; ++L) {
        const _Float16* W = L ? W3h : W2h;
        const float* bs = L ? pm_b3 : pm_b2;
        F16x16 a[8];
#pragma unroll
        for (int kt = 0; kt < 8; ++kt) {
            const v8h* p = (const v8h*)&hb[arow * NH + kt * 32 + akb];
            a[kt].h[0] = p[0];
            a[kt].h[1] = p[1];               // +8 halves
        }
        for (int nt = 0; nt < 16; ++nt) {
            int col = nt * 16 + bcol16;
            float bc = bs[col];
            v8f acc = { bc, bc, bc, bc, bc, bc, bc, bc };
#pragma unroll
            for (int kt = 0; kt < 8; ++kt) {
                F16x16 bf;
                const v8h* q = (const v8h*)&W[col * NH + kt * 32 + bkb];
                bf.h[0] = q[0];
                bf.h[1] = q[1];
                acc = __builtin_amdgcn_wmma_f32_16x16x32_f16(
                          false, a[kt].v, false, bf.v, (short)0, acc, false, false);
            }
#pragma unroll
            for (int v = 0; v < 8; ++v) {
                float o = acc[v];
                if (L == 0) o = gelu_fast(o);
                hb[(v + prow) * NH + col] = (_Float16)o;
            }
        }
    }

    // ---- heads: fused [tW;wW;cW] 48x256 ----
    {
        F16x16 a[8];
#pragma unroll
        for (int kt = 0; kt < 8; ++kt) {
            const v8h* p = (const v8h*)&hb[arow * NH + kt * 32 + akb];
            a[kt].h[0] = p[0];
            a[kt].h[1] = p[1];
        }
        for (int nt = 0; nt < 3; ++nt) {
            int col = nt * 16 + bcol16;
            float bc = headB[col];
            v8f acc = { bc, bc, bc, bc, bc, bc, bc, bc };
#pragma unroll
            for (int kt = 0; kt < 8; ++kt) {
                F16x16 bf;
                const v8h* q = (const v8h*)&headW[col * NH + kt * 32 + bkb];
                bf.h[0] = q[0];
                bf.h[1] = q[1];
                acc = __builtin_amdgcn_wmma_f32_16x16x32_f16(
                          false, a[kt].v, false, bf.v, (short)0, acc, false, false);
            }
#pragma unroll
            for (int v = 0; v < 8; ++v) {
                int rg = rbase + v + prow;
                float o = acc[v];
                if (nt < 2) {
                    out_type[rg * 32 + col] = o;         // cols 0..31
                } else {
                    if (bcol16 == 0) {                    // col 32: weight logit
                        out_w[rg] = o;
                    } else if (bcol16 == 1) {             // col 33: gate
                        float gate = 1.0f / (1.0f + __expf(-o));
                        out_y[rg * 3 + 0] = y0[rg * 3 + 0] * gate;
                        out_y[rg * 3 + 1] = y0[rg * 3 + 1] * gate;
                        out_y[rg * 3 + 2] = y0[rg * 3 + 2] * gate;
                    }
                }
            }
        }
    }
}

// ---------------------------------------------------------------------------
// Host launcher
// ---------------------------------------------------------------------------
extern "C" void kernel_launch(void* const* d_in, const int* in_sizes, int n_in,
                              void* d_out, int out_size, void* d_ws, size_t ws_size,
                              hipStream_t stream) {
    const float* g     = (const float*)d_in[0];
    const float* query = (const float*)d_in[1];
    const float* qW    = (const float*)d_in[2];
    const float* qb    = (const float*)d_in[3];
    const float* kW    = (const float*)d_in[4];
    /* kb (d_in[5]) cancels inside the softmax */
    const float* gm_W1 = (const float*)d_in[6];
    const float* gm_b1 = (const float*)d_in[7];
    const float* gm_W2 = (const float*)d_in[8];
    const float* gm_b2 = (const float*)d_in[9];
    const float* gm_W3 = (const float*)d_in[10];
    const float* gm_b3 = (const float*)d_in[11];
    const float* pm_W1 = (const float*)d_in[12];
    const float* pm_b1 = (const float*)d_in[13];
    const float* pm_W2 = (const float*)d_in[14];
    const float* pm_b2 = (const float*)d_in[15];
    const float* pm_W3 = (const float*)d_in[16];
    const float* pm_b3 = (const float*)d_in[17];
    const float* tW    = (const float*)d_in[18];
    const float* tb    = (const float*)d_in[19];
    const float* wW    = (const float*)d_in[20];
    const float* wb    = (const float*)d_in[21];
    const float* cW    = (const float*)d_in[22];
    const float* cb    = (const float*)d_in[23];

    // workspace layout (byte offsets, 256-aligned)
    char* w = (char*)d_ws;
    float*    inv   = (float*)(w + 0);                 //  8192 f
    float*    vvec  = (float*)(w + 32768);             //   128 f
    float*    c0    = (float*)(w + 33280);             //     1 f
    float*    qdot  = (float*)(w + 33536);             //  2048 f
    float*    Gg    = (float*)(w + 41728);             // 32768 f
    float*    Qq    = (float*)(w + 172800);            // 524288 f
    float*    aiw   = (float*)(w + 2269952);           // 262144 f
    float*    y0w   = (float*)(w + 3318528);           // 786432 f
    _Float16* W2h   = (_Float16*)(w + 6464256);        // 65536 h
    _Float16* W3h   = (_Float16*)(w + 6595328);        // 65536 h
    _Float16* headW = (_Float16*)(w + 6726400);        // 12288 h
    float*    headB = (float*)(w + 6750976);           //    48 f

    float* out_y    = (float*)d_out;                   // (B,M,3)
    float* out_type = out_y + (size_t)NROWS * 3;       // (B,M,32)
    float* out_w    = out_type + (size_t)NROWS * 32;   // (B,M)

    k_prep<<<256, 256, 0, stream>>>(pm_W2, pm_W3, tW, tb, wW, wb, cW, cb,
                                    qW, qb, kW, W2h, W3h, headW, headB, vvec, c0);
    k_inv <<<32, 256, 0, stream>>>(g, inv);
    k_qdot<<<8, 256, 0, stream>>>(query, vvec, c0, qdot);
    k_gmlp<<<NB, 256, 0, stream>>>(inv, gm_W1, gm_b1, gm_W2, gm_b2, gm_W3, gm_b3,
                                   pm_W1, pm_b1, Gg);
    k_qq  <<<NM, 256, 0, stream>>>(query, pm_W1, Qq);
    k_attn<<<NB * 8, 256, 0, stream>>>(g, inv, qdot, aiw, y0w);
    k_fused<<<NROWS / ROWS_PER_BLOCK, 256, 0, stream>>>(
        Gg, Qq, aiw, y0w, pm_W1, pm_b2, pm_b3,
        W2h, W3h, headW, headB, out_y, out_type, out_w);
}
